// READ_41815801594318
// MI455X (gfx1250) — matrix operand order repaired and checked by reading, not compiled, not run on hardware
//
#include <hip/hip_runtime.h>
#include <hip/hip_bf16.h>
#include <math.h>
#include <stdint.h>

#define N_ 2048
#define H_ 256
#define NH_ 8
#define DH_ 32
#define EPS_ 1e-5f
#define ALPHA_ 0.2f
#define BETA_ 0.5f

typedef __bf16 bf16_t;
typedef __attribute__((ext_vector_type(16))) __bf16 v16bf;
typedef __attribute__((ext_vector_type(8)))  __bf16 v8bf;
typedef __attribute__((ext_vector_type(8)))  float  v8f;

// ---------------------------------------------------------------------------
// Generic WMMA GEMM:  C[M,Nn] = act( alpha * A @ B(^T) + bias )
// A: f32 row-major [M,K] (lda). B: f32 [K,Nn] (ldb) or, if transB, [Nn,K] (ldb).
// f32 operands are converted to bf16 while staging into LDS; accumulate in f32
// via v_wmma_f32_16x16x32_bf16. Block tile 128x64, 8 waves, 32x32 per wave.
// Fast path: unguarded float4 loads + packed 16B LDS stores (full k-tile,
// aligned stride). Slow path (edge tiles only): guarded scalar loads.
// act: 0=none 1=relu 2=tanh
// ---------------------------------------------------------------------------
#define BM 128
#define BN 64
#define BK 32

__global__ __launch_bounds__(256)
void gemm_wmma(const float* __restrict__ A, int lda,
               const float* __restrict__ B, int ldb, int transB,
               float* __restrict__ C, int ldc,
               int M, int K, int Nn,
               float alpha, const float* __restrict__ bias, int act)
{
  __shared__ __align__(16) bf16_t As[BM * BK];   // As[r*32 + k]
  __shared__ __align__(16) bf16_t Bs[BN * BK];   // Bs[n*32 + k]  (k-contiguous)

  const int tid  = threadIdx.x;
  const int m0   = blockIdx.y * BM;
  const int n0   = blockIdx.x * BN;
  const int wid  = tid >> 5;
  const int lane = tid & 31;
  const int half = lane >> 4;     // 0 or 1
  const int l16  = lane & 15;
  const int wm   = (wid >> 1) * 32;   // wave row base: 0,32,64,96
  const int wn   = (wid & 1) * 32;    // wave col base: 0,32

  // staging coordinates
  const int ar = tid >> 1;            // 0..127   A row within tile
  const int ac = (tid & 1) * 16;      // 0 or 16  A col chunk
  const int bn = tid >> 2;            // 0..63    B col within tile
  const int bk = (tid & 3) * 8;       // 0,8,16,24

  const bool aAligned = ((lda & 3) == 0) && ((((uintptr_t)A) & 15) == 0);
  const bool bAligned = ((ldb & 3) == 0) && ((((uintptr_t)B) & 15) == 0);

  v8f acc[2][2] = {};

  const int kTiles = (K + BK - 1) / BK;
  for (int kt = 0; kt < kTiles; ++kt) {
    const int k0 = kt * BK;
    const bool fullK = (k0 + BK) <= K;
    __syncthreads();

    // ---- stage A tile (128x32): 2 threads/row, 16 elems each ----
    {
      const int gr = m0 + ar;
      if (fullK && aAligned && gr < M) {
        const float* src = A + (size_t)gr * lda + (k0 + ac);
        #pragma unroll
        for (int j = 0; j < 2; ++j) {
          const float4 f0 = *(const float4*)(src + j * 8);
          const float4 f1 = *(const float4*)(src + j * 8 + 4);
          v8bf p;
          p[0] = (bf16_t)f0.x; p[1] = (bf16_t)f0.y;
          p[2] = (bf16_t)f0.z; p[3] = (bf16_t)f0.w;
          p[4] = (bf16_t)f1.x; p[5] = (bf16_t)f1.y;
          p[6] = (bf16_t)f1.z; p[7] = (bf16_t)f1.w;
          *(v8bf*)&As[ar * BK + ac + j * 8] = p;
        }
        if (k0 + BK < K) __builtin_prefetch(src + BK, 0, 0);   // global_prefetch_b8
      } else {
        #pragma unroll
        for (int e = 0; e < 16; ++e) {
          const int gk = k0 + ac + e;
          const float v = (gr < M && gk < K) ? A[(size_t)gr * lda + gk] : 0.0f;
          As[ar * BK + ac + e] = (bf16_t)v;
        }
      }
    }
    // ---- stage B tile transposed (64 n x 32 k): 4 threads/col, 8 k each ----
    {
      const int gn = n0 + bn;
      if (fullK && gn < Nn) {
        v8bf p;
        if (transB) {
          const float* src = B + (size_t)gn * ldb + (k0 + bk);
          if (bAligned) {
            const float4 f0 = *(const float4*)(src);
            const float4 f1 = *(const float4*)(src + 4);
            p[0] = (bf16_t)f0.x; p[1] = (bf16_t)f0.y;
            p[2] = (bf16_t)f0.z; p[3] = (bf16_t)f0.w;
            p[4] = (bf16_t)f1.x; p[5] = (bf16_t)f1.y;
            p[6] = (bf16_t)f1.z; p[7] = (bf16_t)f1.w;
          } else {
            #pragma unroll
            for (int e = 0; e < 8; ++e) p[e] = (bf16_t)src[e];
          }
          if (k0 + BK < K) __builtin_prefetch(src + BK, 0, 0);
        } else {
          const float* src = B + (size_t)(k0 + bk) * ldb + gn;
          #pragma unroll
          for (int e = 0; e < 8; ++e) p[e] = (bf16_t)src[(size_t)e * ldb];
        }
        *(v8bf*)&Bs[bn * BK + bk] = p;
      } else {
        #pragma unroll
        for (int e = 0; e < 8; ++e) {
          const int gk = k0 + bk + e;
          float v = 0.0f;
          if (gn < Nn && gk < K)
            v = transB ? B[(size_t)gn * ldb + gk] : B[(size_t)gk * ldb + gn];
          Bs[bn * BK + bk + e] = (bf16_t)v;
        }
      }
    }
    __syncthreads();

    // ---- fragments + WMMA ----
    #pragma unroll
    for (int mi = 0; mi < 2; ++mi) {
      // CDNA5 16-bit A 16x32 layout: lanes 0-15 row M=l16 K{0..7,16..23},
      // lanes 16-31 same rows K{8..15,24..31}
      const int arow = wm + mi * 16 + l16;
      v8bf alo = *(const v8bf*)&As[arow * BK + half * 8];
      v8bf ahi = *(const v8bf*)&As[arow * BK + 16 + half * 8];
      v16bf a;
      #pragma unroll
      for (int e = 0; e < 8; ++e) { a[e] = alo[e]; a[8 + e] = ahi[e]; }
      #pragma unroll
      for (int ni = 0; ni < 2; ++ni) {
        // B 32x16: lane holds column N=l16; lanes 0-15 K{0..7,16..23}, 16-31 K{8..15,24..31}
        const int bcol = wn + ni * 16 + l16;
        v8bf b0 = *(const v8bf*)&Bs[bcol * BK + half * 8];
        v8bf b1 = *(const v8bf*)&Bs[bcol * BK + 16 + half * 8];
        v16bf b;
        #pragma unroll
        for (int e = 0; e < 8; ++e) { b[e] = b0[e]; b[8 + e] = b1[e]; }
        acc[mi][ni] = __builtin_amdgcn_wmma_f32_16x16x32_bf16(
            false, a, false, b, (short)0, acc[mi][ni], false, false);
      }
    }
  }
  // ---- epilogue: C/D layout: lane l16 = column, VGPR r = row half*8+r ----
  #pragma unroll
  for (int mi = 0; mi < 2; ++mi) {
    #pragma unroll
    for (int ni = 0; ni < 2; ++ni) {
      const int gcol = n0 + wn + ni * 16 + l16;
      if (gcol >= Nn) continue;
      const float bv = bias ? bias[gcol] : 0.0f;
      #pragma unroll
      for (int r = 0; r < 8; ++r) {
        const int grow = m0 + wm + mi * 16 + half * 8 + r;
        if (grow >= M) continue;
        float v = alpha * acc[mi][ni][r] + bv;
        if (act == 1)      v = v > 0.f ? v : 0.f;
        else if (act == 2) v = tanhf(v);
        C[(size_t)grow * ldc + gcol] = v;
      }
    }
  }
}

// ---------------------------------------------------------------------------
// Elementwise / reduction kernels (block = 256 threads everywhere)
// ---------------------------------------------------------------------------
__global__ void softmax_rows(float* __restrict__ S, int n, float scale) {
  __shared__ float red[256];
  const int row = blockIdx.x;
  float* p = S + (size_t)row * n;
  float mx = -INFINITY;
  for (int j = threadIdx.x; j < n; j += 256) mx = fmaxf(mx, p[j] * scale);
  red[threadIdx.x] = mx; __syncthreads();
  for (int s = 128; s > 0; s >>= 1) { if (threadIdx.x < s) red[threadIdx.x] = fmaxf(red[threadIdx.x], red[threadIdx.x + s]); __syncthreads(); }
  mx = red[0]; __syncthreads();
  float sum = 0.f;
  for (int j = threadIdx.x; j < n; j += 256) { float e = __expf(p[j] * scale - mx); p[j] = e; sum += e; }
  red[threadIdx.x] = sum; __syncthreads();
  for (int s = 128; s > 0; s >>= 1) { if (threadIdx.x < s) red[threadIdx.x] += red[threadIdx.x + s]; __syncthreads(); }
  const float inv = 1.0f / red[0];
  for (int j = threadIdx.x; j < n; j += 256) p[j] *= inv;
}

__global__ void add_layernorm(const float* __restrict__ h, const float* __restrict__ o,
                              const float* __restrict__ g, const float* __restrict__ b,
                              float* __restrict__ out) {
  __shared__ float red[256];
  const int row = blockIdx.x, j = threadIdx.x;
  const size_t idx = (size_t)row * H_ + j;
  const float y = h[idx] + o[idx];
  red[j] = y; __syncthreads();
  for (int s = 128; s > 0; s >>= 1) { if (j < s) red[j] += red[j + s]; __syncthreads(); }
  const float mu = red[0] / H_; __syncthreads();
  const float d = y - mu;
  red[j] = d * d; __syncthreads();
  for (int s = 128; s > 0; s >>= 1) { if (j < s) red[j] += red[j + s]; __syncthreads(); }
  const float var = red[0] / H_;
  out[idx] = d * rsqrtf(var + EPS_) * g[j] + b[j];
}

__global__ void rowsum_dinv(const float* __restrict__ A, float* __restrict__ dinv) {
  __shared__ float red[256];
  const int row = blockIdx.x;
  float s = 0.f;
  for (int j = threadIdx.x; j < N_; j += 256) s += A[(size_t)row * N_ + j];
  red[threadIdx.x] = s; __syncthreads();
  for (int t = 128; t > 0; t >>= 1) { if (threadIdx.x < t) red[threadIdx.x] += red[threadIdx.x + t]; __syncthreads(); }
  if (!threadIdx.x) { const float d = red[0] + 1.0f; dinv[row] = d > 0.f ? rsqrtf(d) : 0.f; }
}

__global__ void rowscale(const float* __restrict__ x, const float* __restrict__ dinv,
                         float* __restrict__ y) {
  const size_t idx = (size_t)blockIdx.x * H_ + threadIdx.x;
  y[idx] = dinv[blockIdx.x] * x[idx];
}

__global__ void gcn_combine(const float* __restrict__ t, const float* __restrict__ hs,
                            const float* __restrict__ dinv, const float* __restrict__ b,
                            float* __restrict__ out) {
  const int row = blockIdx.x, j = threadIdx.x;
  const size_t idx = (size_t)row * H_ + j;
  out[idx] = dinv[row] * (t[idx] + hs[idx]) + b[j];
}

__global__ void bn_stats(const float* __restrict__ x, float* __restrict__ mean,
                         float* __restrict__ var) {
  __shared__ float r1[256]; __shared__ float r2[256];
  const int col = blockIdx.x;
  float s = 0.f, s2 = 0.f;
  for (int i = threadIdx.x; i < N_; i += 256) { const float v = x[(size_t)i * H_ + col]; s += v; s2 += v * v; }
  r1[threadIdx.x] = s; r2[threadIdx.x] = s2; __syncthreads();
  for (int t = 128; t > 0; t >>= 1) { if (threadIdx.x < t) { r1[threadIdx.x] += r1[threadIdx.x + t]; r2[threadIdx.x] += r2[threadIdx.x + t]; } __syncthreads(); }
  if (!threadIdx.x) { const float mu = r1[0] / N_; mean[col] = mu; var[col] = r2[0] / N_ - mu * mu; }
}

__global__ void bn_apply_leaky(const float* __restrict__ x, const float* __restrict__ mean,
                               const float* __restrict__ var, const float* __restrict__ g,
                               const float* __restrict__ b, float* __restrict__ out) {
  const int row = blockIdx.x, j = threadIdx.x;
  const size_t idx = (size_t)row * H_ + j;
  const float v = (x[idx] - mean[j]) * rsqrtf(var[j] + EPS_) * g[j] + b[j];
  out[idx] = v > 0.f ? v : 0.01f * v;
}

__global__ void mha3_attn(const float* __restrict__ q, const float* __restrict__ k,
                          const float* __restrict__ v, float* __restrict__ out) {
  const int idx = blockIdx.x * blockDim.x + threadIdx.x;
  if (idx >= N_ * NH_) return;
  const int n = idx / NH_, hh = idx % NH_;
  const float scale = 0.17677669529663687f;   // 1/sqrt(32)
  float s[3][3];
  for (int l = 0; l < 3; ++l)
    for (int m = 0; m < 3; ++m) {
      const float* qp = q + ((size_t)l * N_ + n) * H_ + hh * DH_;
      const float* kp = k + ((size_t)m * N_ + n) * H_ + hh * DH_;
      float acc = 0.f;
      for (int d = 0; d < DH_; ++d) acc += qp[d] * kp[d];
      s[l][m] = acc * scale;
    }
  for (int l = 0; l < 3; ++l) {
    const float mx = fmaxf(s[l][0], fmaxf(s[l][1], s[l][2]));
    const float e0 = __expf(s[l][0] - mx), e1 = __expf(s[l][1] - mx), e2 = __expf(s[l][2] - mx);
    const float inv = 1.f / (e0 + e1 + e2);
    s[l][0] = e0 * inv; s[l][1] = e1 * inv; s[l][2] = e2 * inv;
  }
  for (int l = 0; l < 3; ++l) {
    float* op = out + ((size_t)l * N_ + n) * H_ + hh * DH_;
    for (int d = 0; d < DH_; ++d) {
      float acc = 0.f;
      for (int m = 0; m < 3; ++m) acc += s[l][m] * v[((size_t)m * N_ + n) * H_ + hh * DH_ + d];
      op[d] = acc;
    }
  }
}

__global__ void mix_outs(const float* __restrict__ attn, const float* __restrict__ views,
                         float* __restrict__ outs, int n) {
  const int i = blockIdx.x * 256 + threadIdx.x;
  if (i < n) outs[i] = ALPHA_ * attn[i] + (1.f - ALPHA_) * views[i];
}

__global__ void rowdot(const float* __restrict__ t, const float* __restrict__ qf,
                       float* __restrict__ r) {
  __shared__ float red[256];
  const int row = blockIdx.x;
  red[threadIdx.x] = t[(size_t)row * H_ + threadIdx.x] * qf[threadIdx.x];
  __syncthreads();
  for (int k = 128; k > 0; k >>= 1) { if (threadIdx.x < k) red[threadIdx.x] += red[threadIdx.x + k]; __syncthreads(); }
  if (!threadIdx.x) r[row] = red[0];
}

__global__ void view_mean(const float* __restrict__ r, float* __restrict__ w) {
  __shared__ float red[256];
  const int v = blockIdx.x;
  float s = 0.f;
  for (int i = threadIdx.x; i < N_; i += 256) s += r[(size_t)v * N_ + i];
  red[threadIdx.x] = s; __syncthreads();
  for (int t = 128; t > 0; t >>= 1) { if (threadIdx.x < t) red[threadIdx.x] += red[threadIdx.x + t]; __syncthreads(); }
  if (!threadIdx.x) w[v] = red[0] / N_;
}

__global__ void beta_softmax(const float* __restrict__ w, float* __restrict__ beta) {
  const float m  = fmaxf(w[0], fmaxf(w[1], w[2]));
  const float e0 = __expf(w[0] - m), e1 = __expf(w[1] - m), e2 = __expf(w[2] - m);
  const float inv = 1.f / (e0 + e1 + e2);
  beta[0] = e0 * inv; beta[1] = e1 * inv; beta[2] = e2 * inv;
}

__global__ void fused_combine(const float* __restrict__ outs, const float* __restrict__ beta,
                              float* __restrict__ fused) {
  const size_t idx = (size_t)blockIdx.x * H_ + threadIdx.x;
  const size_t V = (size_t)N_ * H_;
  fused[idx] = beta[0] * outs[idx] + beta[1] * outs[V + idx] + beta[2] * outs[2 * V + idx];
}

__global__ void out3_kernel(const float* __restrict__ views, const float* __restrict__ fused,
                            float* __restrict__ out3) {
  const int i = blockIdx.x, j = threadIdx.x;            // i in [0,3N)
  const size_t idx = (size_t)i * H_ + j;
  const int n = i % N_;
  out3[idx] = BETA_ * views[idx] + (1.f - BETA_) * fused[(size_t)n * H_ + j];
}

__global__ void attr_partial(const float* __restrict__ ip, const float* __restrict__ adj,
                             float* __restrict__ psum, float* __restrict__ pcnt) {
  __shared__ float r1[256]; __shared__ float r2[256];
  const size_t total = (size_t)N_ * N_;
  float s = 0.f, c = 0.f;
  for (size_t i = (size_t)blockIdx.x * 256 + threadIdx.x; i < total; i += (size_t)gridDim.x * 256) {
    const float a = adj[i];
    if (a != 0.f) { const float d = ip[i] - a; s += d * d; c += 1.f; }
  }
  r1[threadIdx.x] = s; r2[threadIdx.x] = c; __syncthreads();
  for (int t = 128; t > 0; t >>= 1) { if (threadIdx.x < t) { r1[threadIdx.x] += r1[threadIdx.x + t]; r2[threadIdx.x] += r2[threadIdx.x + t]; } __syncthreads(); }
  if (!threadIdx.x) { psum[blockIdx.x] = r1[0]; pcnt[blockIdx.x] = r2[0]; }
}

__global__ void attr_final(const float* __restrict__ psum, const float* __restrict__ pcnt,
                           int n, float* __restrict__ out) {
  __shared__ float r1[256]; __shared__ float r2[256];
  float s = 0.f, c = 0.f;
  for (int i = threadIdx.x; i < n; i += 256) { s += psum[i]; c += pcnt[i]; }
  r1[threadIdx.x] = s; r2[threadIdx.x] = c; __syncthreads();
  for (int t = 128; t > 0; t >>= 1) { if (threadIdx.x < t) { r1[threadIdx.x] += r1[threadIdx.x + t]; r2[threadIdx.x] += r2[threadIdx.x + t]; } __syncthreads(); }
  if (!threadIdx.x) out[0] = r1[0] / fmaxf(r2[0], 1.f);
}

__global__ void rowlse(const float* __restrict__ S, float* __restrict__ rmax,
                       float* __restrict__ rlse) {
  __shared__ float red[256];
  const int row = blockIdx.x;
  const float* p = S + (size_t)row * N_;
  float mx = -INFINITY;
  for (int j = threadIdx.x; j < N_; j += 256) mx = fmaxf(mx, p[j]);
  red[threadIdx.x] = mx; __syncthreads();
  for (int s = 128; s > 0; s >>= 1) { if (threadIdx.x < s) red[threadIdx.x] = fmaxf(red[threadIdx.x], red[threadIdx.x + s]); __syncthreads(); }
  mx = red[0]; __syncthreads();
  float sum = 0.f;
  for (int j = threadIdx.x; j < N_; j += 256) sum += __expf(p[j] - mx);
  red[threadIdx.x] = sum; __syncthreads();
  for (int s = 128; s > 0; s >>= 1) { if (threadIdx.x < s) red[threadIdx.x] += red[threadIdx.x + s]; __syncthreads(); }
  if (!threadIdx.x) { rmax[row] = mx; rlse[row] = logf(red[0]); }
}

__global__ void mob_partial(const float* __restrict__ S, const float* __restrict__ mob,
                            const float* __restrict__ rmax, const float* __restrict__ rlse,
                            int transM, float* __restrict__ psum) {
  __shared__ float red[256];
  const size_t total = (size_t)N_ * N_;
  float s = 0.f;
  for (size_t idx = (size_t)blockIdx.x * 256 + threadIdx.x; idx < total; idx += (size_t)gridDim.x * 256) {
    const int i = (int)(idx / N_), j = (int)(idx % N_);
    const float m = transM ? mob[(size_t)j * N_ + i] : mob[idx];
    if (m != 0.f) s += -m * (S[idx] - rmax[i] - rlse[i]);
  }
  red[threadIdx.x] = s; __syncthreads();
  for (int t = 128; t > 0; t >>= 1) { if (threadIdx.x < t) red[threadIdx.x] += red[threadIdx.x + t]; __syncthreads(); }
  if (!threadIdx.x) psum[blockIdx.x] = red[0];
}

__global__ void reduce_sum(const float* __restrict__ in, int n, float* __restrict__ out) {
  __shared__ float red[256];
  float s = 0.f;
  for (int i = threadIdx.x; i < n; i += 256) s += in[i];
  red[threadIdx.x] = s; __syncthreads();
  for (int t = 128; t > 0; t >>= 1) { if (threadIdx.x < t) red[threadIdx.x] += red[threadIdx.x + t]; __syncthreads(); }
  if (!threadIdx.x) *out = red[0];
}

__global__ void final_write(const float* __restrict__ scal, float* __restrict__ out) {
  out[0] = scal[0] + scal[1] + scal[2];
}

// ---------------------------------------------------------------------------
// Host orchestration
// ---------------------------------------------------------------------------
static inline void launch_gemm(hipStream_t st, const float* A, int lda,
                               const float* B, int ldb, int tB,
                               float* C, int ldc, int M, int K, int Nn,
                               float alpha, const float* bias, int act) {
  dim3 grid((Nn + BN - 1) / BN, (M + BM - 1) / BM);
  gemm_wmma<<<grid, 256, 0, st>>>(A, lda, B, ldb, tB, C, ldc, M, K, Nn, alpha, bias, act);
}

// Flat param indices (JAX pytree order: dict keys sorted; 'W..' < lowercase)
enum {
  IN_VA = 0, IN_VI = 1, IN_VO = 2, IN_ADJ = 3, IN_MOB = 4,
  IN_BNB0 = 5, IN_BNB1 = 6, IN_BNG0 = 7, IN_BNG1 = 8,
  IN_DYNA_W1 = 9, IN_DYNA_W2 = 10, IN_DYNI_W1 = 11, IN_DYNI_W2 = 12,
  IN_DYNO_W1 = 13, IN_DYNO_W2 = 14,
  IN_ENCA = 15, IN_ENCI = 24, IN_ENCO = 33,   // each: Wk,Wo,Wp,Wq,Wv,b,bo,bp,g
  IN_FUS_WF = 42, IN_FUS_BF = 43, IN_FUS_QF = 44,
  IN_GCNB0 = 45, IN_GCNB1 = 46, IN_GCNB2 = 47,
  IN_GCNW0 = 48, IN_GCNW1 = 49, IN_GCNW2 = 50,
  IN_MHA_WK = 51, IN_MHA_WO = 52, IN_MHA_WQ = 53, IN_MHA_WV = 54,
  IN_MHA_BK = 55, IN_MHA_BO = 56, IN_MHA_BQ = 57, IN_MHA_BV = 58
};
enum { E_WK = 0, E_WO = 1, E_WP = 2, E_WQ = 3, E_WV = 4, E_B = 5, E_BO = 6, E_BP = 7, E_G = 8 };

struct Ws {
  float *S, *Aa, *Ai, *Ao, *views, *h, *q, *k, *v, *o, *t1, *t2, *t3;
  float *q3, *k3, *v3, *ao3, *big3a, *big3b;
  float *dinvA, *dinvI, *dinvO, *rmax, *rlse, *rdot, *mean, *var, *psum, *pcnt, *w3, *beta3, *scal;
};

static void encoder_fwd(hipStream_t st, const float* x, int din,
                        void* const* din_ptrs, int base, Ws& W, float* outv) {
  const float* Wk = (const float*)din_ptrs[base + E_WK];
  const float* Wo = (const float*)din_ptrs[base + E_WO];
  const float* Wp = (const float*)din_ptrs[base + E_WP];
  const float* Wq = (const float*)din_ptrs[base + E_WQ];
  const float* Wv = (const float*)din_ptrs[base + E_WV];
  const float* b  = (const float*)din_ptrs[base + E_B];
  const float* bo = (const float*)din_ptrs[base + E_BO];
  const float* bp = (const float*)din_ptrs[base + E_BP];
  const float* g  = (const float*)din_ptrs[base + E_G];

  launch_gemm(st, x, din, Wp, H_, 0, W.h, H_, N_, din, H_, 1.f, bp, 0);      // h = x@Wp+bp
  launch_gemm(st, W.h, H_, Wq, H_, 0, W.q, H_, N_, H_, H_, 1.f, nullptr, 0);
  launch_gemm(st, W.h, H_, Wk, H_, 0, W.k, H_, N_, H_, H_, 1.f, nullptr, 0);
  launch_gemm(st, W.h, H_, Wv, H_, 0, W.v, H_, N_, H_, H_, 1.f, nullptr, 0);
  const float scale = 0.17677669529663687f;   // 1/sqrt(DH)
  for (int hh = 0; hh < NH_; ++hh) {
    launch_gemm(st, W.q + hh * DH_, H_, W.k + hh * DH_, H_, 1,
                W.S, N_, N_, DH_, N_, 1.f, nullptr, 0);                      // S = Qh @ Kh^T
    softmax_rows<<<N_, 256, 0, st>>>(W.S, N_, scale);
    launch_gemm(st, W.S, N_, W.v + hh * DH_, H_, 0,
                W.o + hh * DH_, H_, N_, N_, DH_, 1.f, nullptr, 0);           // Oh = P @ Vh
  }
  launch_gemm(st, W.o, H_, Wo, H_, 0, W.t1, H_, N_, H_, H_, 1.f, bo, 0);     // o2 = o@Wo+bo
  add_layernorm<<<N_, 256, 0, st>>>(W.h, W.t1, g, b, outv);                  // LN(h+o2)
}

static void gcn_layer(hipStream_t st, float* vb, const float* A, const float* dinv,
                      const float* Wm, const float* bvec, Ws& W,
                      const float* bnG, const float* bnB, bool last) {
  launch_gemm(st, vb, H_, Wm, H_, 0, W.t1, H_, N_, H_, H_, 1.f, nullptr, 0);   // hW = v@W
  rowscale<<<N_, 256, 0, st>>>(W.t1, dinv, W.t2);                               // hs = dinv*hW
  launch_gemm(st, A, N_, W.t2, H_, 0, W.t1, H_, N_, N_, H_, 1.f, nullptr, 0);  // t = A@hs
  if (last) {
    gcn_combine<<<N_, 256, 0, st>>>(W.t1, W.t2, dinv, bvec, vb);
  } else {
    gcn_combine<<<N_, 256, 0, st>>>(W.t1, W.t2, dinv, bvec, W.t3);
    bn_stats<<<H_, 256, 0, st>>>(W.t3, W.mean, W.var);
    bn_apply_leaky<<<N_, 256, 0, st>>>(W.t3, W.mean, W.var, bnG, bnB, vb);
  }
}

extern "C" void kernel_launch(void* const* d_in, const int* in_sizes, int n_in,
                              void* d_out, int out_size, void* d_ws, size_t ws_size,
                              hipStream_t stream) {
  (void)in_sizes; (void)n_in; (void)out_size; (void)ws_size;
  const size_t NN  = (size_t)N_ * N_;
  const size_t NHf = (size_t)N_ * H_;
  float* P = (float*)d_ws;
  Ws W; size_t off = 0;
  W.S     = P + off; off += NN;
  W.Aa    = P + off; off += NN;
  W.Ai    = P + off; off += NN;
  W.Ao    = P + off; off += NN;
  W.views = P + off; off += 3 * NHf;
  W.h  = P + off; off += NHf;  W.q  = P + off; off += NHf;
  W.k  = P + off; off += NHf;  W.v  = P + off; off += NHf;
  W.o  = P + off; off += NHf;  W.t1 = P + off; off += NHf;
  W.t2 = P + off; off += NHf;  W.t3 = P + off; off += NHf;
  W.q3 = P + off; off += 3 * NHf;  W.k3 = P + off; off += 3 * NHf;
  W.v3 = P + off; off += 3 * NHf;  W.ao3 = P + off; off += 3 * NHf;
  W.big3a = P + off; off += 3 * NHf;  W.big3b = P + off; off += 3 * NHf;
  W.dinvA = P + off; off += N_;  W.dinvI = P + off; off += N_;
  W.dinvO = P + off; off += N_;
  W.rmax = P + off; off += N_;  W.rlse = P + off; off += N_;
  W.rdot = P + off; off += 3 * N_;
  W.mean = P + off; off += H_;  W.var = P + off; off += H_;
  W.psum = P + off; off += 1024;  W.pcnt = P + off; off += 1024;
  W.w3 = P + off; off += 4;  W.beta3 = P + off; off += 4;  W.scal = P + off; off += 16;

  const float* adj = (const float*)d_in[IN_ADJ];
  const float* mob = (const float*)d_in[IN_MOB];
  float* va = W.views;
  float* vi = W.views + NHf;
  float* vo = W.views + 2 * NHf;

  // ---- encoders ----
  encoder_fwd(stream, (const float*)d_in[IN_VA], 9,   d_in, IN_ENCA, W, va);
  encoder_fwd(stream, (const float*)d_in[IN_VI], 270, d_in, IN_ENCI, W, vi);
  encoder_fwd(stream, (const float*)d_in[IN_VO], 270, d_in, IN_ENCO, W, vo);

  // ---- dynamic adjacencies: A = relu((vW1)(vW2)^T) ----
  const float* viewp[3] = { va, vi, vo };
  float* Amat[3] = { W.Aa, W.Ai, W.Ao };
  float* dinvp[3] = { W.dinvA, W.dinvI, W.dinvO };
  const int dynW1[3] = { IN_DYNA_W1, IN_DYNI_W1, IN_DYNO_W1 };
  const int dynW2[3] = { IN_DYNA_W2, IN_DYNI_W2, IN_DYNO_W2 };
  for (int vv = 0; vv < 3; ++vv) {
    launch_gemm(stream, viewp[vv], H_, (const float*)d_in[dynW1[vv]], H_, 0,
                W.t1, H_, N_, H_, H_, 1.f, nullptr, 0);
    launch_gemm(stream, viewp[vv], H_, (const float*)d_in[dynW2[vv]], H_, 0,
                W.t2, H_, N_, H_, H_, 1.f, nullptr, 0);
    launch_gemm(stream, W.t1, H_, W.t2, H_, 1, Amat[vv], N_,
                N_, H_, N_, 1.f, nullptr, 1 /*relu*/);
    rowsum_dinv<<<N_, 256, 0, stream>>>(Amat[vv], dinvp[vv]);
  }

  // ---- GCN layers ----
  const int gW[3] = { IN_GCNW0, IN_GCNW1, IN_GCNW2 };
  const int gB[3] = { IN_GCNB0, IN_GCNB1, IN_GCNB2 };
  const int bG[2] = { IN_BNG0, IN_BNG1 };
  const int bB[2] = { IN_BNB0, IN_BNB1 };
  for (int layer = 0; layer < 3; ++layer) {
    const bool last = (layer == 2);
    for (int vv = 0; vv < 3; ++vv) {
      gcn_layer(stream, (float*)viewp[vv], Amat[vv], dinvp[vv],
                (const float*)d_in[gW[layer]], (const float*)d_in[gB[layer]], W,
                last ? nullptr : (const float*)d_in[bG[layer]],
                last ? nullptr : (const float*)d_in[bB[layer]], last);
    }
  }

  // ---- mha3 over the 3 views ----
  launch_gemm(stream, W.views, H_, (const float*)d_in[IN_MHA_WQ], H_, 0,
              W.q3, H_, 3 * N_, H_, H_, 1.f, (const float*)d_in[IN_MHA_BQ], 0);
  launch_gemm(stream, W.views, H_, (const float*)d_in[IN_MHA_WK], H_, 0,
              W.k3, H_, 3 * N_, H_, H_, 1.f, (const float*)d_in[IN_MHA_BK], 0);
  launch_gemm(stream, W.views, H_, (const float*)d_in[IN_MHA_WV], H_, 0,
              W.v3, H_, 3 * N_, H_, H_, 1.f, (const float*)d_in[IN_MHA_BV], 0);
  mha3_attn<<<(N_ * NH_ + 255) / 256, 256, 0, stream>>>(W.q3, W.k3, W.v3, W.ao3);
  launch_gemm(stream, W.ao3, H_, (const float*)d_in[IN_MHA_WO], H_, 0,
              W.big3b, H_, 3 * N_, H_, H_, 1.f, (const float*)d_in[IN_MHA_BO], 0);
  mix_outs<<<(int)((3 * NHf + 255) / 256), 256, 0, stream>>>(W.big3b, W.views, W.big3a, (int)(3 * NHf));

  // ---- fusion ----
  launch_gemm(stream, W.big3a, H_, (const float*)d_in[IN_FUS_WF], H_, 0,
              W.big3b, H_, 3 * N_, H_, H_, 1.f, (const float*)d_in[IN_FUS_BF], 2 /*tanh*/);
  rowdot<<<3 * N_, 256, 0, stream>>>(W.big3b, (const float*)d_in[IN_FUS_QF], W.rdot);
  view_mean<<<3, 256, 0, stream>>>(W.rdot, W.w3);
  beta_softmax<<<1, 1, 0, stream>>>(W.w3, W.beta3);
  fused_combine<<<N_, 256, 0, stream>>>(W.big3a, W.beta3, W.t1);     // fused (N,H)
  out3_kernel<<<3 * N_, 256, 0, stream>>>(W.views, W.t1, W.big3b);   // out (3,N,H)

  const float* out0 = W.big3b;
  const float* sM   = W.big3b + NHf;
  const float* dM   = W.big3b + 2 * NHf;

  // ---- attr loss: masked MSE over ip = out0 @ out0^T ----
  launch_gemm(stream, out0, H_, out0, H_, 1, W.S, N_, N_, H_, N_, 1.f, nullptr, 0);
  attr_partial<<<1024, 256, 0, stream>>>(W.S, adj, W.psum, W.pcnt);
  attr_final<<<1, 256, 0, stream>>>(W.psum, W.pcnt, 1024, W.scal + 0);

  // ---- mob loss part 1: -mob * log_softmax(s @ d^T) ----
  launch_gemm(stream, sM, H_, dM, H_, 1, W.S, N_, N_, H_, N_, 1.f, nullptr, 0);
  rowlse<<<N_, 256, 0, stream>>>(W.S, W.rmax, W.rlse);
  mob_partial<<<1024, 256, 0, stream>>>(W.S, mob, W.rmax, W.rlse, 0, W.psum);
  reduce_sum<<<1, 256, 0, stream>>>(W.psum, 1024, W.scal + 1);

  // ---- mob loss part 2: -mob^T * log_softmax(d @ s^T) ----
  launch_gemm(stream, dM, H_, sM, H_, 1, W.S, N_, N_, H_, N_, 1.f, nullptr, 0);
  rowlse<<<N_, 256, 0, stream>>>(W.S, W.rmax, W.rlse);
  mob_partial<<<1024, 256, 0, stream>>>(W.S, mob, W.rmax, W.rlse, 1, W.psum);
  reduce_sum<<<1, 256, 0, stream>>>(W.psum, 1024, W.scal + 2);

  final_write<<<1, 1, 0, stream>>>(W.scal, (float*)d_out);
}
